// MultiHeadSelfAttention_51419348468339
// MI455X (gfx1250) — compile-verified, matrix operand order checked by reading, not run on hardware
//
#include <hip/hip_runtime.h>
#include <math.h>

// ---------------------------------------------------------------------------
// MHA with per-head full-D projections, B=4 L=1024 D=1024 H=16.
// ~825 GFLOP vs ~24us of HBM traffic at 23.3TB/s => compute-bound; all four
// matmul stages run through v_wmma_f32_16x16x32_bf16 with f32 accumulation.
// All GEMMs are NT (B stored [N,K]); v is produced pre-transposed and Wo is
// transposed once, so every LDS store is a b128 and the inner loop is a
// single-barrier double-buffered software pipeline with 8 WMMAs per K-step.
// ---------------------------------------------------------------------------

typedef __attribute__((ext_vector_type(16))) __bf16 v16bf;
typedef __attribute__((ext_vector_type(8)))  float  v8f;

static constexpr int Bc = 4, Lc = 1024, Dc = 1024, Hc = 16;
static constexpr long long BLD  = (long long)Bc * Lc * Dc;        // 4,194,304
static constexpr long long DD   = (long long)Dc * Dc;             // 1,048,576
static constexpr long long HDD  = (long long)Hc * DD;             // 16,777,216
static constexpr long long HBLD = (long long)Hc * BLD;            // 67,108,864
static constexpr long long BHLL = (long long)Bc * Hc * Lc * Lc;   // 67,108,864

#define TM 128
#define TN 64
#define TK 32
#define NTHREADS 128   // 4 wave32s; each wave owns a 32x64 strip

// ---------------------------------------------------------------------------
// f32 -> bf16 cast (grid-stride)
// ---------------------------------------------------------------------------
__global__ void cast_f32_bf16(const float* __restrict__ src,
                              __bf16* __restrict__ dst, long long n) {
    long long i = (long long)blockIdx.x * blockDim.x + threadIdx.x;
    long long stride = (long long)gridDim.x * blockDim.x;
    for (; i < n; i += stride) dst[i] = (__bf16)src[i];
}

// ---------------------------------------------------------------------------
// Tiled transpose + cast: src f32 [rows, cols] -> dst bf16 [cols, rows].
// 32x32 tiles through LDS (padded to kill bank conflicts), 256 threads.
// ---------------------------------------------------------------------------
__global__ void __launch_bounds__(256)
transpose_cast(const float* __restrict__ src, __bf16* __restrict__ dst,
               int rows, int cols) {
    __shared__ float tile[32][33];
    const int r0 = blockIdx.y * 32, c0 = blockIdx.x * 32;
    const int tx = threadIdx.x & 31, ty = threadIdx.x >> 5;
#pragma unroll
    for (int i = ty; i < 32; i += 8)
        tile[i][tx] = src[(size_t)(r0 + i) * cols + c0 + tx];
    __syncthreads();
#pragma unroll
    for (int i = ty; i < 32; i += 8)
        dst[(size_t)(c0 + i) * rows + r0 + tx] = (__bf16)tile[tx][i];
}

// ---------------------------------------------------------------------------
// 128x64 NT GEMM core, double-buffered LDS, one barrier per K-step.
//   A : [M,K] row-major, pre-offset to the block's 128-row strip
//   Bm: [N,K] row-major, pre-offset to the block's 64-row strip
// Each wave: 32(M) x 64(N) strip = acc[2][4], 8 WMMAs per 32-wide K-step.
// ---------------------------------------------------------------------------
#define SA_BUF (TM * TK)   // 4096 bf16 elems per buffer
#define SB_BUF (TN * TK)   // 2048 bf16 elems per buffer

__device__ __forceinline__ void gemm_tile_nt(const __bf16* __restrict__ A, int lda,
                                             const __bf16* __restrict__ Bm, int ldb,
                                             int K, v8f acc[2][4],
                                             __bf16* sA, __bf16* sB) {
    const int t    = threadIdx.x;
    const int wave = t >> 5;
    const int lane = t & 31;
    const int mrow = lane & 15;          // fragment row within 16
    const int ksel = (lane >> 4) << 4;   // lanes 0-15 -> K 0..15, 16-31 -> 16..31

    // cooperative staging: thread t covers A rows (t>>1) and (t>>1)+64,
    // B row (t>>1); 16 contiguous bf16 (32B) per segment.
    const int    arow  = t >> 1;
    const int    ahalf = (t & 1) << 4;
    const size_t aoff0 = (size_t)arow * lda + ahalf;
    const size_t aoff1 = (size_t)(arow + 64) * lda + ahalf;
    const size_t boff  = (size_t)arow * ldb + ahalf;
    const int    sa0   = arow * TK + ahalf;
    const int    sa1   = (arow + 64) * TK + ahalf;
    const int    sb0   = arow * TK + ahalf;

    // ---- prologue: stage K-tile 0 into buffer 0 ----
    v16bf a0 = *(const v16bf*)&A[aoff0];
    v16bf a1 = *(const v16bf*)&A[aoff1];
    v16bf b0 = *(const v16bf*)&Bm[boff];
    *(v16bf*)&sA[sa0] = a0;
    *(v16bf*)&sA[sa1] = a1;
    *(v16bf*)&sB[sb0] = b0;
    __syncthreads();

    for (int k0 = 0; k0 < K; k0 += TK) {
        const int  cur  = (k0 >> 5) & 1;
        const int  nxt  = cur ^ 1;
        const bool more = (k0 + TK) < K;

        // issue global loads for the next K-tile before computing this one
        if (more) {
            a0 = *(const v16bf*)&A[aoff0 + k0 + TK];
            a1 = *(const v16bf*)&A[aoff1 + k0 + TK];
            b0 = *(const v16bf*)&Bm[boff + k0 + TK];
            __builtin_prefetch(&A[aoff0 + k0 + 2 * TK], 0, 1);
            __builtin_prefetch(&A[aoff1 + k0 + 2 * TK], 0, 1);
            __builtin_prefetch(&Bm[boff + k0 + 2 * TK], 0, 1);
        }

        // compute current tile: 2 A fragments x 4 B fragments = 8 WMMAs
        const __bf16* cA = sA + cur * SA_BUF;
        const __bf16* cB = sB + cur * SB_BUF;
        v16bf af0 = *(const v16bf*)&cA[(wave * 32      + mrow) * TK + ksel];
        v16bf af1 = *(const v16bf*)&cA[(wave * 32 + 16 + mrow) * TK + ksel];
#pragma unroll
        for (int ns = 0; ns < 4; ++ns) {
            v16bf bfr = *(const v16bf*)&cB[(ns * 16 + mrow) * TK + ksel];
            acc[0][ns] = __builtin_amdgcn_wmma_f32_16x16x32_bf16(
                false, af0, false, bfr, (short)0, acc[0][ns], false, false);
            acc[1][ns] = __builtin_amdgcn_wmma_f32_16x16x32_bf16(
                false, af1, false, bfr, (short)0, acc[1][ns], false, false);
        }

        // stage next tile into the alternate buffer
        if (more) {
            *(v16bf*)&sA[nxt * SA_BUF + sa0] = a0;
            *(v16bf*)&sA[nxt * SA_BUF + sa1] = a1;
            *(v16bf*)&sB[nxt * SB_BUF + sb0] = b0;
        }
        __syncthreads();
    }
}

#define DECLARE_LDS()                                           \
    __shared__ __align__(16) __bf16 sA[2 * SA_BUF];             \
    __shared__ __align__(16) __bf16 sB[2 * SB_BUF];

#define ZERO_ACC()                                              \
    v8f acc[2][4];                                              \
    _Pragma("unroll") for (int _m = 0; _m < 2; ++_m)            \
    _Pragma("unroll") for (int _n = 0; _n < 4; ++_n)            \
    _Pragma("unroll") for (int _i = 0; _i < 8; ++_i)            \
        acc[_m][_n][_i] = 0.0f;

// C/D layout: element i of v8f -> row = i + 8*(lane>=16), col = lane&15.
// Variadic so commas inside the body don't split macro arguments.
#define EPILOGUE_LOOP(...)                                                \
    {                                                                     \
        const int lane  = threadIdx.x & 31;                               \
        const int wave  = threadIdx.x >> 5;                               \
        const int colb  = lane & 15;                                      \
        const int rhalf = (lane >> 4) << 3;                               \
        _Pragma("unroll")                                                 \
        for (int mi = 0; mi < 2; ++mi) {                                  \
            _Pragma("unroll")                                             \
            for (int ns = 0; ns < 4; ++ns) {                              \
                const int col = n0 + ns * 16 + colb;                      \
                _Pragma("unroll")                                         \
                for (int i = 0; i < 8; ++i) {                             \
                    const int row = m0 + wave * 32 + mi * 16 + rhalf + i; \
                    __VA_ARGS__                                           \
                }                                                         \
            }                                                             \
        }                                                                 \
    }

// ---------------------------------------------------------------------------
// Stage 1: q/k/v projections.  z = proj*H + h.  NT: x_bf [4096,1024] x
// W[h][N=e rows, K=d].  q,k stored [H,B,L,D]; v stored transposed [H,B,D,L]
// so the ctx GEMM is NT too.
// ---------------------------------------------------------------------------
__global__ void __launch_bounds__(NTHREADS)
gemm_qkv(const __bf16* __restrict__ xb, const __bf16* __restrict__ wb,
         __bf16* __restrict__ qkv) {
    DECLARE_LDS();
    const int proj = blockIdx.z >> 4;
    const int h    = blockIdx.z & 15;
    const int m0   = blockIdx.y * TM;
    const int n0   = blockIdx.x * TN;

    ZERO_ACC();
    const __bf16* A  = xb + (size_t)m0 * Dc;
    const __bf16* Bm = wb + (size_t)proj * HDD + (size_t)h * DD + (size_t)n0 * Dc;
    gemm_tile_nt(A, Dc, Bm, Dc, Dc, acc, sA, sB);

    __bf16* outp = qkv + (size_t)proj * HBLD + (size_t)h * BLD;
    if (proj < 2) {
        EPILOGUE_LOOP({ outp[(size_t)row * Dc + col] = (__bf16)acc[mi][ns][i]; })
    } else {
        // vT[h][b][d][l]: row = b*L + l, col = d
        EPILOGUE_LOOP({
            const int b = row >> 10;
            const int l = row & 1023;
            outp[(size_t)b * Dc * Lc + (size_t)col * Lc + l] = (__bf16)acc[mi][ns][i];
        })
    }
}

// ---------------------------------------------------------------------------
// Stage 2: logits = q.k^T / sqrt(D) - 1e9*mask -> f32 [B,H,L,L] in d_out.
// z = h*B + b.  NT (both q,k rows are [L,D]).
// ---------------------------------------------------------------------------
__global__ void __launch_bounds__(NTHREADS)
gemm_logits(const __bf16* __restrict__ qkv, const float* __restrict__ mask,
            float* __restrict__ attn) {
    DECLARE_LDS();
    const int h  = blockIdx.z >> 2;
    const int b  = blockIdx.z & 3;
    const int m0 = blockIdx.y * TM;
    const int n0 = blockIdx.x * TN;

    ZERO_ACC();
    const __bf16* A  = qkv + (size_t)h * BLD + (size_t)b * Lc * Dc + (size_t)m0 * Dc;
    const __bf16* Bm = qkv + HBLD + (size_t)h * BLD + (size_t)b * Lc * Dc + (size_t)n0 * Dc;
    gemm_tile_nt(A, Dc, Bm, Dc, Dc, acc, sA, sB);

    float* outp = attn + ((size_t)b * Hc + h) * Lc * Lc;
    const float* mb = mask + (size_t)b * Lc * Lc;
    EPILOGUE_LOOP({
        outp[(size_t)row * Lc + col] =
            acc[mi][ns][i] * 0.03125f - 1.0e9f * mb[(size_t)row * Lc + col];
    })
}

// ---------------------------------------------------------------------------
// Stage 3: row softmax over L=1024, in place in d_out (f32) + bf16 copy of P.
// ---------------------------------------------------------------------------
__global__ void __launch_bounds__(256)
softmax_rows(float* __restrict__ attn, __bf16* __restrict__ pbf) {
    __shared__ float red[256];
    const size_t row = blockIdx.x;
    float* rp = attn + row * Lc;
    __bf16* pp = pbf + row * Lc;
    const int tid = threadIdx.x;

    float v[4], m = -3.4e38f;
#pragma unroll
    for (int i = 0; i < 4; ++i) { v[i] = rp[tid + i * 256]; m = fmaxf(m, v[i]); }
    red[tid] = m; __syncthreads();
    for (int s = 128; s > 0; s >>= 1) {
        if (tid < s) red[tid] = fmaxf(red[tid], red[tid + s]);
        __syncthreads();
    }
    m = red[0]; __syncthreads();

    float sum = 0.f;
#pragma unroll
    for (int i = 0; i < 4; ++i) { v[i] = __expf(v[i] - m); sum += v[i]; }
    red[tid] = sum; __syncthreads();
    for (int s = 128; s > 0; s >>= 1) {
        if (tid < s) red[tid] += red[tid + s];
        __syncthreads();
    }
    const float inv = 1.0f / red[0];
#pragma unroll
    for (int i = 0; i < 4; ++i) {
        float w = v[i] * inv;
        rp[tid + i * 256] = w;
        pp[tid + i * 256] = (__bf16)w;
    }
}

// ---------------------------------------------------------------------------
// Stage 4: ctx = P.v using vT [H,B,D,L] (NT).  Epilogue performs the torch
// cat-of-views permutation into "combined" [B,L,H*D].  z = h*B + b.
// ---------------------------------------------------------------------------
__global__ void __launch_bounds__(NTHREADS)
gemm_ctx(const __bf16* __restrict__ pbf, const __bf16* __restrict__ qkv,
         __bf16* __restrict__ comb) {
    DECLARE_LDS();
    const int h  = blockIdx.z >> 2;
    const int b  = blockIdx.z & 3;
    const int m0 = blockIdx.y * TM;
    const int n0 = blockIdx.x * TN;

    ZERO_ACC();
    const __bf16* A  = pbf + ((size_t)b * Hc + h) * Lc * Lc + (size_t)m0 * Lc;
    const __bf16* Bm = qkv + 2 * HBLD + (size_t)h * BLD + (size_t)b * Dc * Lc
                       + (size_t)n0 * Lc;   // vT rows are d, cols are l
    gemm_tile_nt(A, Lc, Bm, Lc, Lc, acc, sA, sB);

    EPILOGUE_LOOP({
        const int j = (((col >> 6) * Hc + h) << 6) + (col & 63);
        comb[((size_t)b * Lc + row) * (size_t)(Hc * Dc) + j] = (__bf16)acc[mi][ns][i];
    })
}

// ---------------------------------------------------------------------------
// Stage 5: context = combined @ Wo via WoT [D, H*D] (NT), f32 into d_out.
// ---------------------------------------------------------------------------
__global__ void __launch_bounds__(NTHREADS)
gemm_out(const __bf16* __restrict__ comb, const __bf16* __restrict__ woT,
         float* __restrict__ ctx) {
    DECLARE_LDS();
    const int m0 = blockIdx.y * TM;
    const int n0 = blockIdx.x * TN;

    ZERO_ACC();
    const __bf16* A  = comb + (size_t)m0 * (Hc * Dc);
    const __bf16* Bm = woT + (size_t)n0 * (Hc * Dc);
    gemm_tile_nt(A, Hc * Dc, Bm, Hc * Dc, Hc * Dc, acc, sA, sB);

    EPILOGUE_LOOP({ ctx[(size_t)row * Dc + col] = acc[mi][ns][i]; })
}

// ---------------------------------------------------------------------------
extern "C" void kernel_launch(void* const* d_in, const int* in_sizes, int n_in,
                              void* d_out, int out_size, void* d_ws, size_t ws_size,
                              hipStream_t stream) {
    (void)in_sizes; (void)n_in; (void)out_size; (void)ws_size;
    const float* x    = (const float*)d_in[0];
    const float* mask = (const float*)d_in[1];
    const float* Wq   = (const float*)d_in[2];
    const float* Wk   = (const float*)d_in[3];
    const float* Wv   = (const float*)d_in[4];
    const float* Wo   = (const float*)d_in[5];

    float* ctx_out  = (float*)d_out;          // [B,L,D]
    float* attn_out = ctx_out + BLD;          // [B,H,L,L]

    // workspace carve-up (all bf16)
    __bf16* wsb     = (__bf16*)d_ws;
    __bf16* x_bf    = wsb;                    size_t off = BLD;
    __bf16* w_bf    = wsb + off;              off += 3 * HDD;   // Wq,Wk,Wv
    __bf16* woT_bf  = wsb + off;              off += HDD;       // Wo transposed
    __bf16* qkv_bf  = wsb + off;              off += 3 * HBLD;  // q,k,vT
    __bf16* p_bf    = wsb + off;              off += BHLL;
    __bf16* comb_bf = wsb + off;

    // 1) casts (+ one-shot transpose of Wo to [D, H*D])
    cast_f32_bf16<<<1024, 256, 0, stream>>>(x,  x_bf,         BLD);
    cast_f32_bf16<<<2048, 256, 0, stream>>>(Wq, w_bf,         HDD);
    cast_f32_bf16<<<2048, 256, 0, stream>>>(Wk, w_bf + HDD,   HDD);
    cast_f32_bf16<<<2048, 256, 0, stream>>>(Wv, w_bf + 2*HDD, HDD);
    transpose_cast<<<dim3(Dc / 32, (Hc * Dc) / 32), 256, 0, stream>>>(
        Wo, woT_bf, Hc * Dc, Dc);

    // 2) q/k/v projections: grid (N/64, M/128, 3*H)
    gemm_qkv<<<dim3(Dc / TN, (Bc * Lc) / TM, 3 * Hc), NTHREADS, 0, stream>>>(
        x_bf, w_bf, qkv_bf);

    // 3) masked logits -> d_out attention region
    gemm_logits<<<dim3(Lc / TN, Lc / TM, Hc * Bc), NTHREADS, 0, stream>>>(
        qkv_bf, mask, attn_out);

    // 4) softmax rows (B*H*L blocks), f32 in place + bf16 P
    softmax_rows<<<Bc * Hc * Lc, 256, 0, stream>>>(attn_out, p_bf);

    // 5) ctx = P.vT with cat-of-views permutation into combined layout
    gemm_ctx<<<dim3(Dc / TN, Lc / TM, Hc * Bc), NTHREADS, 0, stream>>>(
        p_bf, qkv_bf, comb_bf);

    // 6) context = combined @ WoT
    gemm_out<<<dim3(Dc / TN, (Bc * Lc) / TM, 1), NTHREADS, 0, stream>>>(
        comb_bf, woT_bf, ctx_out);
}
